// GraphConvolution_39926015983992
// MI455X (gfx1250) — matrix operand order repaired and checked by reading, not compiled, not run on hardware
//
#include <hip/hip_runtime.h>
#include <math.h>

// ---------------------------------------------------------------------------
// EGNN layer fused for MI455X (gfx1250, wave32, WMMA f32_16x16x32_f16).
// Compute-bound (~131 GFLOP vs ~1 GB traffic). Edge kernel uses a 32-edge
// M tile with 2x2 register blocking so every B fragment feeds two WMMAs,
// halving VMEM issue per matrix op vs the 16-edge version.
// ---------------------------------------------------------------------------

typedef _Float16 f16;
typedef __attribute__((ext_vector_type(16))) _Float16 v16h;
typedef __attribute__((ext_vector_type(8)))  _Float16 v8h;
typedef __attribute__((ext_vector_type(8)))  float    v8f;

#define DD   256
#define DD2  512
#define INV_NORM 0.01f

__device__ __forceinline__ float sigmoidf_(float x) { return 1.0f / (1.0f + __expf(-x)); }
__device__ __forceinline__ float siluf_(float x)    { return x * sigmoidf_(x); }
__device__ __forceinline__ v8f   zero8()            { v8f z = {0.f,0.f,0.f,0.f,0.f,0.f,0.f,0.f}; return z; }

// A fragment: 16x32 f16 tile (row-major in LDS, leading dim ldk).
// lane m = l&15 ; lanes 0-15 hold K = kb+[0..15]; lanes 16-31 K = kb+[16..31]
__device__ __forceinline__ v16h load_a_frag(const f16* tile, int ldk, int lane, int kb) {
    const int m  = lane & 15;
    const int kh = lane >> 4;
    const f16* p = tile + m * ldk + kb + kh * 16;
    v8h lo = *(const v8h*)p;
    v8h hi = *(const v8h*)(p + 8);
    v16h r;
#pragma unroll
    for (int i = 0; i < 8; ++i) { r[i] = lo[i]; r[i + 8] = hi[i]; }
    return r;
}

// B fragment: 32x16 f16, weights pre-transposed to [N][K] row-major.
// lane n = l&15 ; lanes 0-15 hold K = kb+[0..15], lanes 16-31 K = kb+[16..31]
__device__ __forceinline__ v16h load_b_frag(const f16* Wt, int K, int nbase, int lane, int kb) {
    const int n  = lane & 15;
    const int kh = lane >> 4;
    return *(const v16h*)(Wt + (size_t)(nbase + n) * K + kb + kh * 16);
}

#define WMMA(a, b, c) __builtin_amdgcn_wmma_f32_16x16x32_f16(false, (a), false, (b), (short)0, (c), false, false)

// ------------------------------ prep kernels -------------------------------

__global__ void cvt_f16_kernel(const float* __restrict__ src, f16* __restrict__ dst, int n) {
    int i = blockIdx.x * 256 + threadIdx.x;
    if (i < n) dst[i] = (f16)src[i];
}

// W: [K][Ncols] row-major f32  ->  Wt: [Ncols][K] f16
__global__ void transpose_f16_kernel(const float* __restrict__ W, f16* __restrict__ Wt,
                                     int K, int Ncols) {
    int i = blockIdx.x * 256 + threadIdx.x;
    if (i < K * Ncols) {
        int n = i / K, k = i % K;
        Wt[i] = (f16)W[(size_t)k * Ncols + n];
    }
}

__global__ void copy_f32_kernel(const float* __restrict__ src, float* __restrict__ dst, int n) {
    int i = blockIdx.x * 256 + threadIdx.x;
    if (i < n) dst[i] = src[i];
}

__global__ void zero_f32_kernel(float* __restrict__ p, int n) {
    int i = blockIdx.x * 256 + threadIdx.x;
    if (i < n) p[i] = 0.f;
}

// ------------------------------ edge kernel --------------------------------
// 32 edges / block, 256 threads = 8 waves; each wave owns 32 output columns
// and both 16-row M tiles (2x2 register blocking -> 4 accumulators).
__global__ __launch_bounds__(256) void egnn_edge_kernel(
    const f16*   __restrict__ h16,
    const float* __restrict__ distances,
    const int*   __restrict__ edges,
    const float* __restrict__ edge_mask,
    const float* __restrict__ h_gauss,
    const f16*   __restrict__ eW1t, const float* __restrict__ eW1g,
    const float* __restrict__ eb1,
    const f16*   __restrict__ eW2t, const float* __restrict__ eb2,
    const float* __restrict__ aW,   const float* __restrict__ ab,
    float*       __restrict__ agg,
    int E)
{
    __shared__ f16   xA[32 * DD2];   // 32 KB: [h_row | h_col] tile (32 edges)
    __shared__ f16   hid[32 * DD];   // 16 KB: silu(x@eW1+b) as f16
    __shared__ float gaussS[32];
    __shared__ float scaleS[32];
    __shared__ float attS[32];
    __shared__ int   rowS[32], colS[32];

    const int tid  = threadIdx.x;
    const int lane = tid & 31;
    const int wave = tid >> 5;
    const int e0   = blockIdx.x * 32;

    // stage 1: per-edge scalars (gauss kernel already masked, matches reference x)
    if (tid < 32) {
        int e = e0 + tid;
        float g = 0.f, em = 0.f;
        int r = 0, c = 0;
        if (e < E) {
            float hg = log1pf(__expf(h_gauss[0]));                 // softplus
            float d  = distances[e];
            em = edge_mask[e];
            g  = (0.3989422804014327f / hg) * __expf(-(d * d) / (2.f * hg * hg)) * em;
            r = edges[e];
            c = edges[E + e];
        }
        rowS[tid] = r; colS[tid] = c;
        gaussS[tid] = g;
        scaleS[tid] = em;    // holds edge_mask until attention epilogue
        attS[tid]   = 0.f;
    }
    __syncthreads();

    // stage 2: gather x = [h[row], h[col]] (f16) into LDS. 32 edges x 16 segs of 32 halfs.
#pragma unroll
    for (int s = tid; s < 32 * 16; s += 256) {
        int e   = s >> 4;
        int seg = s & 15;
        const f16* src = (seg < 8) ? (h16 + (size_t)rowS[e] * DD + seg * 32)
                                   : (h16 + (size_t)colS[e] * DD + (seg - 8) * 32);
        f16* dst = xA + e * DD2 + seg * 32;
#pragma unroll
        for (int i = 0; i < 4; ++i)
            *(v8h*)(dst + i * 8) = *(const v8h*)(src + i * 8);
    }
    __syncthreads();

    const int nb0 = wave * 32;
    const int nb1 = wave * 32 + 16;
    const int n    = lane & 15;
    const int half = lane >> 4;
    const int c0 = nb0 + n, c1 = nb1 + n;

    // stage 3: GEMM1, 2x2 blocked: each B fragment feeds two WMMAs
    v8f a00 = zero8(), a01 = zero8(), a10 = zero8(), a11 = zero8();
    for (int kb = 0; kb < DD2; kb += 32) {
        v16h A0 = load_a_frag(xA,             DD2, lane, kb);
        v16h A1 = load_a_frag(xA + 16 * DD2,  DD2, lane, kb);
        v16h B0 = load_b_frag(eW1t, DD2, nb0, lane, kb);
        v16h B1 = load_b_frag(eW1t, DD2, nb1, lane, kb);
        a00 = WMMA(A0, B0, a00);
        a01 = WMMA(A0, B1, a01);
        a10 = WMMA(A1, B0, a10);
        a11 = WMMA(A1, B1, a11);
    }
    {   // epilogue: + gauss rank-1 + bias + SiLU -> hid (f16)
        const float w0 = eW1g[c0], w1 = eW1g[c1];
        const float b0 = eb1[c0],  b1 = eb1[c1];
#pragma unroll
        for (int r = 0; r < 8; ++r) {
            int m0 = r + 8 * half;
            int m1 = m0 + 16;
            float g0 = gaussS[m0], g1 = gaussS[m1];
            hid[m0 * DD + c0] = (f16)siluf_(a00[r] + g0 * w0 + b0);
            hid[m0 * DD + c1] = (f16)siluf_(a01[r] + g0 * w1 + b1);
            hid[m1 * DD + c0] = (f16)siluf_(a10[r] + g1 * w0 + b0);
            hid[m1 * DD + c1] = (f16)siluf_(a11[r] + g1 * w1 + b1);
        }
    }
    __syncthreads();

    // stage 4: GEMM2 -> mij stays in the 4 accumulators (+ bias folded in)
    a00 = zero8(); a01 = zero8(); a10 = zero8(); a11 = zero8();
    for (int kb = 0; kb < DD; kb += 32) {
        v16h A0 = load_a_frag(hid,            DD, lane, kb);
        v16h A1 = load_a_frag(hid + 16 * DD,  DD, lane, kb);
        v16h B0 = load_b_frag(eW2t, DD, nb0, lane, kb);
        v16h B1 = load_b_frag(eW2t, DD, nb1, lane, kb);
        a00 = WMMA(A0, B0, a00);
        a01 = WMMA(A0, B1, a01);
        a10 = WMMA(A1, B0, a10);
        a11 = WMMA(A1, B1, a11);
    }
    {   // fold eb2 into the accumulators: accs now hold mij
        const float b0 = eb2[c0], b1 = eb2[c1];
#pragma unroll
        for (int r = 0; r < 8; ++r) {
            a00[r] += b0; a01[r] += b1;
            a10[r] += b0; a11[r] += b1;
        }
    }

    // stage 5: attention matvec from registers.
    // Lanes within each 16-lane half hold the same rows, different columns:
    // xor-shuffle reduce over masks {1,2,4,8}, then lane (l&15)==0 commits.
    {
        const float w0 = aW[c0], w1 = aW[c1];
#pragma unroll
        for (int r = 0; r < 8; ++r) {
            float p0 = a00[r] * w0 + a01[r] * w1;   // tile 0, row r+8*half
            float p1 = a10[r] * w0 + a11[r] * w1;   // tile 1, row 16+r+8*half
#pragma unroll
            for (int mk = 1; mk < 16; mk <<= 1) {
                p0 += __shfl_xor(p0, mk, 32);
                p1 += __shfl_xor(p1, mk, 32);
            }
            if ((lane & 15) == 0) {
                atomicAdd(&attS[r + 8 * half],      p0);
                atomicAdd(&attS[16 + r + 8 * half], p1);
            }
        }
    }
    __syncthreads();
    if (tid < 32)
        scaleS[tid] = sigmoidf_(attS[tid] + ab[0]) * scaleS[tid] * INV_NORM;
    __syncthreads();

    // stage 6: scatter-add edge features to agg[row] straight from registers
#pragma unroll
    for (int r = 0; r < 8; ++r) {
        int m0 = r + 8 * half;
        int m1 = m0 + 16;
        float s0 = scaleS[m0], s1 = scaleS[m1];
        size_t row0 = (size_t)rowS[m0] * DD;
        size_t row1 = (size_t)rowS[m1] * DD;
        atomicAdd(&agg[row0 + c0], a00[r] * s0);
        atomicAdd(&agg[row0 + c1], a01[r] * s0);
        atomicAdd(&agg[row1 + c0], a10[r] * s1);
        atomicAdd(&agg[row1 + c1], a11[r] * s1);
    }
}

// ------------------------------ node kernel --------------------------------
// 16 nodes / block: out = (h + MLP([h|agg/100])) @ lW + lb
__global__ __launch_bounds__(256) void egnn_node_kernel(
    const float* __restrict__ h,
    const f16*   __restrict__ h16,
    const float* __restrict__ agg,
    const f16*   __restrict__ nW1t, const float* __restrict__ nb1,
    const f16*   __restrict__ nW2t, const float* __restrict__ nb2,
    const f16*   __restrict__ lWt,  const float* __restrict__ lb,
    float*       __restrict__ out, int Nn)
{
    __shared__ f16 zA[16 * DD2];   // 16 KB
    __shared__ f16 t1[16 * DD];    //  8 KB
    __shared__ f16 oS[16 * DD];    //  8 KB

    const int tid  = threadIdx.x;
    const int lane = tid & 31;
    const int wave = tid >> 5;
    const int n0   = blockIdx.x * 16;

    // stage 1: z = [h | agg/100] as f16
    {
        int i    = tid >> 4;
        int seg  = tid & 15;
        int node = n0 + i; if (node >= Nn) node = Nn - 1;
        f16* dst = zA + i * DD2 + seg * 32;
        if (seg < 8) {
            const f16* src = h16 + (size_t)node * DD + seg * 32;
#pragma unroll
            for (int j = 0; j < 4; ++j) *(v8h*)(dst + j * 8) = *(const v8h*)(src + j * 8);
        } else {
            const float* src = agg + (size_t)node * DD + (seg - 8) * 32;
#pragma unroll
            for (int j = 0; j < 32; ++j) dst[j] = (f16)(src[j] * INV_NORM);
        }
    }
    __syncthreads();

    const int nb0 = wave * 32;
    const int nb1c = wave * 32 + 16;

    // GEMM-A: t1 = silu(z @ nW1 + nb1), K = 512
    v8f acc0 = zero8(), acc1 = zero8();
    for (int kb = 0; kb < DD2; kb += 32) {
        v16h a  = load_a_frag(zA, DD2, lane, kb);
        v16h b0 = load_b_frag(nW1t, DD2, nb0, lane, kb);
        v16h b1 = load_b_frag(nW1t, DD2, nb1c, lane, kb);
        acc0 = WMMA(a, b0, acc0);
        acc1 = WMMA(a, b1, acc1);
    }
    {
        const int n = lane & 15, half = lane >> 4;
        const int c0 = nb0 + n, c1 = nb1c + n;
        const float b0 = nb1[c0], b1 = nb1[c1];
#pragma unroll
        for (int r = 0; r < 8; ++r) {
            int m = r + 8 * half;
            t1[m * DD + c0] = (f16)siluf_(acc0[r] + b0);
            t1[m * DD + c1] = (f16)siluf_(acc1[r] + b1);
        }
    }
    __syncthreads();

    // GEMM-B: o = h + t1 @ nW2 + nb2, K = 256 (residual in f32)
    acc0 = zero8(); acc1 = zero8();
    for (int kb = 0; kb < DD; kb += 32) {
        v16h a  = load_a_frag(t1, DD, lane, kb);
        v16h b0 = load_b_frag(nW2t, DD, nb0, lane, kb);
        v16h b1 = load_b_frag(nW2t, DD, nb1c, lane, kb);
        acc0 = WMMA(a, b0, acc0);
        acc1 = WMMA(a, b1, acc1);
    }
    {
        const int n = lane & 15, half = lane >> 4;
        const int c0 = nb0 + n, c1 = nb1c + n;
        const float b0 = nb2[c0], b1 = nb2[c1];
#pragma unroll
        for (int r = 0; r < 8; ++r) {
            int m = r + 8 * half;
            int node = n0 + m; if (node >= Nn) node = Nn - 1;
            float o0 = h[(size_t)node * DD + c0] + acc0[r] + b0;
            float o1 = h[(size_t)node * DD + c1] + acc1[r] + b1;
            oS[m * DD + c0] = (f16)o0;
            oS[m * DD + c1] = (f16)o1;
        }
    }
    __syncthreads();

    // GEMM-C: hidden = o @ lW + lb, K = 256, write f32 output
    acc0 = zero8(); acc1 = zero8();
    for (int kb = 0; kb < DD; kb += 32) {
        v16h a  = load_a_frag(oS, DD, lane, kb);
        v16h b0 = load_b_frag(lWt, DD, nb0, lane, kb);
        v16h b1 = load_b_frag(lWt, DD, nb1c, lane, kb);
        acc0 = WMMA(a, b0, acc0);
        acc1 = WMMA(a, b1, acc1);
    }
    {
        const int n = lane & 15, half = lane >> 4;
        const int c0 = nb0 + n, c1 = nb1c + n;
        const float b0 = lb[c0], b1 = lb[c1];
#pragma unroll
        for (int r = 0; r < 8; ++r) {
            int m = r + 8 * half;
            int node = n0 + m;
            if (node < Nn) {
                out[(size_t)node * DD + c0] = acc0[r] + b0;
                out[(size_t)node * DD + c1] = acc1[r] + b1;
            }
        }
    }
}

// ------------------------------ launcher -----------------------------------

extern "C" void kernel_launch(void* const* d_in, const int* in_sizes, int n_in,
                              void* d_out, int out_size, void* d_ws, size_t ws_size,
                              hipStream_t stream) {
    const float* h         = (const float*)d_in[0];
    const float* distances = (const float*)d_in[1];
    const int*   edges     = (const int*)d_in[2];
    // d_in[3] = node_mask (unused by reference math)
    const float* edge_mask = (const float*)d_in[4];
    const float* h_gauss   = (const float*)d_in[5];
    const float* eW1 = (const float*)d_in[6];
    const float* eb1 = (const float*)d_in[7];
    const float* eW2 = (const float*)d_in[8];
    const float* eb2 = (const float*)d_in[9];
    const float* aW  = (const float*)d_in[10];
    const float* ab  = (const float*)d_in[11];
    const float* nW1 = (const float*)d_in[12];
    const float* nb1 = (const float*)d_in[13];
    const float* nW2 = (const float*)d_in[14];
    const float* nb2 = (const float*)d_in[15];
    const float* lW  = (const float*)d_in[16];
    const float* lb  = (const float*)d_in[17];

    const int Nn = in_sizes[0] / DD;   // 10000
    const int E  = in_sizes[1];        // 320000

    // workspace carve-out (~16.3 MB)
    char*  ws  = (char*)d_ws;
    size_t off = 0;
    auto alloc = [&](size_t bytes) -> void* {
        void* p = ws + off;
        off += (bytes + 255) & ~(size_t)255;
        return p;
    };
    f16*   h16  = (f16*)  alloc((size_t)Nn * DD * sizeof(f16));
    f16*   eW1t = (f16*)  alloc((size_t)DD * DD2 * sizeof(f16));
    float* eW1g = (float*)alloc(DD * sizeof(float));
    f16*   eW2t = (f16*)  alloc((size_t)DD * DD * sizeof(f16));
    f16*   nW1t = (f16*)  alloc((size_t)DD * DD2 * sizeof(f16));
    f16*   nW2t = (f16*)  alloc((size_t)DD * DD * sizeof(f16));
    f16*   lWt  = (f16*)  alloc((size_t)DD * DD * sizeof(f16));
    float* agg  = (float*)alloc((size_t)Nn * DD * sizeof(float));

    // prep (cheap, serialized on stream)
    {
        int nh = Nn * DD;
        cvt_f16_kernel<<<(nh + 255) / 256, 256, 0, stream>>>(h, h16, nh);
        transpose_f16_kernel<<<(DD2 * DD + 255) / 256, 256, 0, stream>>>(eW1, eW1t, DD2, DD);
        copy_f32_kernel<<<1, 256, 0, stream>>>(eW1 + (size_t)DD2 * DD, eW1g, DD);
        transpose_f16_kernel<<<(DD * DD + 255) / 256, 256, 0, stream>>>(eW2, eW2t, DD, DD);
        transpose_f16_kernel<<<(DD2 * DD + 255) / 256, 256, 0, stream>>>(nW1, nW1t, DD2, DD);
        transpose_f16_kernel<<<(DD * DD + 255) / 256, 256, 0, stream>>>(nW2, nW2t, DD, DD);
        transpose_f16_kernel<<<(DD * DD + 255) / 256, 256, 0, stream>>>(lW, lWt, DD, DD);
        zero_f32_kernel<<<(Nn * DD + 255) / 256, 256, 0, stream>>>(agg, Nn * DD);
    }

    // fused edge MLP + attention + scatter (dominant ~126 GFLOP)
    egnn_edge_kernel<<<(E + 31) / 32, 256, 0, stream>>>(
        h16, distances, edges, edge_mask, h_gauss,
        eW1t, eW1g, eb1, eW2t, eb2, aW, ab, agg, E);

    // fused node MLP + residual + final linear
    egnn_node_kernel<<<(Nn + 15) / 16, 256, 0, stream>>>(
        h, h16, agg, nW1t, nb1, nW2t, nb2, lWt, lb, (float*)d_out, Nn);
}
